// QuantumAttentionHead2025_21638045237785
// MI455X (gfx1250) — compile-verified, matrix operand order checked by reading, not run on hardware
//
#include <hip/hip_runtime.h>
#include <hip/hip_bf16.h>

typedef _Float16 v16h __attribute__((ext_vector_type(16)));
typedef _Float16 v8h  __attribute__((ext_vector_type(8)));
typedef float    v8f  __attribute__((ext_vector_type(8)));

#define HIDDEN 1024
#define S_LEN  2048
#define BATCH  4
#define HEAD   64
#define NQ     8
#define ROWS   8192   /* BATCH * S_LEN */

// 16-bit fragment layout (CDNA5 ISA 7.12.2): lane half h holds K runs
// [k0+8h .. k0+8h+7] in frag[0..7] and [k0+16+8h .. +7] in frag[8..15].
__device__ __forceinline__ v16h mk_frag(v8h lo, v8h hi) {
    v16h f;
#pragma unroll
    for (int i = 0; i < 8; ++i) { f[i] = lo[i]; f[8 + i] = hi[i]; }
    return f;
}

__device__ __forceinline__ v16h load_frag_f16(const _Float16* p, int half) {
    const v8h lo = *reinterpret_cast<const v8h*>(p + 8 * half);
    const v8h hi = *reinterpret_cast<const v8h*>(p + 16 + 8 * half);
    return mk_frag(lo, hi);
}

// ---------------------------------------------------------------------------
// Kernel 0: one-time fp32 -> f16 conversion of the three weight matrices.
// ---------------------------------------------------------------------------
__global__ void __launch_bounds__(256) wcvt_kernel(
    const float* __restrict__ Wq, const float* __restrict__ Wk,
    const float* __restrict__ Wv, _Float16* __restrict__ Wqh,
    _Float16* __restrict__ Wkh, _Float16* __restrict__ Wvh)
{
    const int i = blockIdx.x * blockDim.x + threadIdx.x;   // 65536 / 4 threads
    if (i * 4 >= HEAD * HIDDEN) return;
    const float4 q = *reinterpret_cast<const float4*>(Wq + i * 4);
    const float4 k = *reinterpret_cast<const float4*>(Wk + i * 4);
    const float4 v = *reinterpret_cast<const float4*>(Wv + i * 4);
    _Float16 oq[4] = {(_Float16)q.x, (_Float16)q.y, (_Float16)q.z, (_Float16)q.w};
    _Float16 ok[4] = {(_Float16)k.x, (_Float16)k.y, (_Float16)k.z, (_Float16)k.w};
    _Float16 ov[4] = {(_Float16)v.x, (_Float16)v.y, (_Float16)v.z, (_Float16)v.w};
    *reinterpret_cast<ulong1*>(Wqh + i * 4) = *reinterpret_cast<ulong1*>(oq);
    *reinterpret_cast<ulong1*>(Wkh + i * 4) = *reinterpret_cast<ulong1*>(ok);
    *reinterpret_cast<ulong1*>(Wvh + i * 4) = *reinterpret_cast<ulong1*>(ov);
}

// ---------------------------------------------------------------------------
// Kernel 1: fused Q/K/V projections.  [8192,1024] x [1024,64]^T (+bias)
// Output: Qh, Kh row-major f16 [row][64];  Vt dim-major f16 [b][64][S].
// ---------------------------------------------------------------------------
__global__ void __launch_bounds__(128) qkv_project_kernel(
    const float* __restrict__ query, const float* __restrict__ key,
    const float* __restrict__ value,
    const _Float16* __restrict__ Wqh, const float* __restrict__ bq,
    const _Float16* __restrict__ Wkh, const float* __restrict__ bk,
    const _Float16* __restrict__ Wvh, const float* __restrict__ bv,
    _Float16* __restrict__ Qh, _Float16* __restrict__ Kh,
    _Float16* __restrict__ Vt)
{
    const int tid  = threadIdx.x;
    const int wave = tid >> 5;
    const int lane = tid & 31;
    const int half = lane >> 4;
    const int ln   = lane & 15;
    const int wrow0 = blockIdx.x * 64 + wave * 16;   // 16 rows per wave

    v8f accQ[4] = {}, accK[4] = {}, accV[4] = {};

    for (int k0 = 0; k0 < HIDDEN; k0 += 32) {
        // A fragments: 16 rows x 32 K of each input, fp32 -> f16 (required cvt)
        const long arow = (long)(wrow0 + ln) * HIDDEN + k0;
        v16h aq, ak, av;
#pragma unroll
        for (int g = 0; g < 2; ++g) {                 // two 8-float runs
            const long off = arow + (g ? 16 : 0) + 8 * half;
#pragma unroll
            for (int p = 0; p < 2; ++p) {             // two float4 per run
                const float4 fq = *reinterpret_cast<const float4*>(query + off + 4 * p);
                const float4 fk = *reinterpret_cast<const float4*>(key   + off + 4 * p);
                const float4 fv = *reinterpret_cast<const float4*>(value + off + 4 * p);
                const int e = g * 8 + p * 4;
                aq[e+0]=(_Float16)fq.x; aq[e+1]=(_Float16)fq.y; aq[e+2]=(_Float16)fq.z; aq[e+3]=(_Float16)fq.w;
                ak[e+0]=(_Float16)fk.x; ak[e+1]=(_Float16)fk.y; ak[e+2]=(_Float16)fk.z; ak[e+3]=(_Float16)fk.w;
                av[e+0]=(_Float16)fv.x; av[e+1]=(_Float16)fv.y; av[e+2]=(_Float16)fv.z; av[e+3]=(_Float16)fv.w;
            }
        }
#pragma unroll
        for (int a = 0; a < 4; ++a) {
            const long wrow = (long)(a * 16 + ln) * HIDDEN + k0;
            const v16h bqf = load_frag_f16(Wqh + wrow, half);
            const v16h bkf = load_frag_f16(Wkh + wrow, half);
            const v16h bvf = load_frag_f16(Wvh + wrow, half);
            accQ[a] = __builtin_amdgcn_wmma_f32_16x16x32_f16(false, aq, false, bqf, (short)0, accQ[a], false, false);
            accK[a] = __builtin_amdgcn_wmma_f32_16x16x32_f16(false, ak, false, bkf, (short)0, accK[a], false, false);
            accV[a] = __builtin_amdgcn_wmma_f32_16x16x32_f16(false, av, false, bvf, (short)0, accV[a], false, false);
        }
    }

    // Bias add + store (C layout: vgpr r -> M = r or r+8 by lane half; lane -> N)
#pragma unroll
    for (int a = 0; a < 4; ++a) {
        const int n = a * 16 + ln;
        const float bqv = bq[n], bkv2 = bk[n], bvv2 = bv[n];
#pragma unroll
        for (int r = 0; r < 8; ++r) {
            const int mloc = half ? (r + 8) : r;
            const int row  = wrow0 + mloc;
            const int bb   = row >> 11;           // / 2048
            const int ss   = row & (S_LEN - 1);
            Qh[(long)row * HEAD + n] = (_Float16)(accQ[a][r] + bqv);
            Kh[(long)row * HEAD + n] = (_Float16)(accK[a][r] + bkv2);
            Vt[((long)bb * HEAD + n) * S_LEN + ss] = (_Float16)(accV[a][r] + bvv2);
        }
    }
}

// ---------------------------------------------------------------------------
// Kernel 2: qubit projection + cosine normalization (eps-clamped), f16 out.
// ---------------------------------------------------------------------------
__global__ void __launch_bounds__(256) c2q_kernel(
    const _Float16* __restrict__ Qh, const _Float16* __restrict__ Kh,
    const float* __restrict__ Wc2q, const float* __restrict__ bc2q,
    _Float16* __restrict__ qnh, _Float16* __restrict__ knh)
{
    const int row = blockIdx.x * blockDim.x + threadIdx.x;
    if (row >= ROWS) return;
    for (int t = 0; t < 2; ++t) {
        const _Float16* src = t ? Kh : Qh;
        _Float16*       dst = t ? knh : qnh;
        float x[HEAD];
#pragma unroll
        for (int g = 0; g < 8; ++g) {
            const v8h v = *reinterpret_cast<const v8h*>(src + (long)row * HEAD + g * 8);
#pragma unroll
            for (int j = 0; j < 8; ++j) x[g * 8 + j] = (float)v[j];
        }
        float y[NQ]; float nrm = 0.f;
#pragma unroll
        for (int q = 0; q < NQ; ++q) {
            float acc = bc2q[q];
#pragma unroll
            for (int d = 0; d < HEAD; ++d) acc += x[d] * Wc2q[q * HEAD + d];
            y[q] = acc; nrm += acc * acc;
        }
        const float inv = 1.f / fmaxf(sqrtf(nrm), 1e-8f);
        _Float16 o[NQ];
#pragma unroll
        for (int q = 0; q < NQ; ++q) o[q] = (_Float16)(y[q] * inv);
        *reinterpret_cast<v8h*>(dst + (long)row * NQ) = *reinterpret_cast<v8h*>(o);
    }
}

// ---------------------------------------------------------------------------
// Kernel 3: flash attention with fused quantum-fidelity scores.
// Block = 128 threads (4 waves), 64 query rows per block, tiles of 64 keys.
// ---------------------------------------------------------------------------
__global__ void __launch_bounds__(128) attn_kernel(
    const _Float16* __restrict__ Qh, const _Float16* __restrict__ Kh,
    const _Float16* __restrict__ Vt, const _Float16* __restrict__ qnh,
    const _Float16* __restrict__ knh, float* __restrict__ out)
{
    __shared__ float    s_s[64 * 64];     // fp32 score tile
    __shared__ _Float16 p_s[64 * 64];     // f16 prob tile (A-layout staging)
    __shared__ float    m_s[64], l_s[64], coef_s[64];

    const int tid = threadIdx.x, wave = tid >> 5, lane = tid & 31;
    const int half = lane >> 4, ln = lane & 15;
    const int b  = blockIdx.x >> 5;               // 32 m-tiles per batch
    const int m0 = (blockIdx.x & 31) * 64;
    const int wrow = m0 + wave * 16;
    const long rowbase = (long)b * S_LEN;

    if (tid < 64) { m_s[tid] = -1e30f; l_s[tid] = 0.f; }
    __syncthreads();

    // Q fragments for K=64 (two 16x32 chunks), loaded once
    v16h qf[2];
#pragma unroll
    for (int kf = 0; kf < 2; ++kf)
        qf[kf] = load_frag_f16(Qh + (rowbase + wrow + ln) * HEAD + kf * 32, half);

    // qn fragment: K=8 real (one v8h for lanes 0-15 / frag[0..7]), zero pad
    v16h qnf = {};
    if (half == 0) {
        const v8h t = *reinterpret_cast<const v8h*>(qnh + (rowbase + wrow + ln) * NQ);
#pragma unroll
        for (int i = 0; i < 8; ++i) qnf[i] = t[i];
    }

    v8f O[4] = {};

    for (int j = 0; j < 32; ++j) {
        const int n0 = j * 64;
        if (j < 31) {   // pull next K/V tiles toward the WGP while we compute
            __builtin_prefetch(Kh + (rowbase + n0 + 64 + lane * 2) * HEAD, 0, 0);
            __builtin_prefetch(Vt + ((long)b * HEAD + lane) * S_LEN + n0 + 64, 0, 0);
        }
        // ----- scores: classical QK^T + quantum fidelity -----
#pragma unroll
        for (int a = 0; a < 4; ++a) {
            const long krow = (rowbase + n0 + a * 16 + ln) * HEAD;
            v8f sacc = {};
#pragma unroll
            for (int kf = 0; kf < 2; ++kf) {
                const v16h kbf = load_frag_f16(Kh + krow + kf * 32, half);
                sacc = __builtin_amdgcn_wmma_f32_16x16x32_f16(false, qf[kf], false, kbf, (short)0, sacc, false, false);
            }
            v16h knf = {};
            if (half == 0) {
                const v8h t = *reinterpret_cast<const v8h*>(knh + (rowbase + n0 + a * 16 + ln) * NQ);
#pragma unroll
                for (int i = 0; i < 8; ++i) knf[i] = t[i];
            }
            v8f facc = {};
            facc = __builtin_amdgcn_wmma_f32_16x16x32_f16(false, qnf, false, knf, (short)0, facc, false, false);
#pragma unroll
            for (int r = 0; r < 8; ++r) {
                const float f  = facc[r];
                const float qw = f * (1.f + 0.1f * __sinf(f * 3.14159265358979f));
                const float sc = 0.5f * qw + 0.0625f * sacc[r];   // 0.5 / sqrt(64)
                const int mloc = half ? (r + 8) : r;
                s_s[(wave * 16 + mloc) * 64 + a * 16 + ln] = sc;
            }
        }
        __syncthreads();
        // ----- online softmax (one thread per row) -----
        if (tid < 64) {
            const int row = tid;
            float mx = -1e30f;
#pragma unroll 8
            for (int n = 0; n < 64; ++n) mx = fmaxf(mx, s_s[row * 64 + n]);
            const float m_old = m_s[row];
            const float m_new = fmaxf(m_old, mx);
            const float cf    = __expf(m_old - m_new);
            float sum = 0.f;
#pragma unroll 8
            for (int n = 0; n < 64; ++n) {
                const float p = __expf(s_s[row * 64 + n] - m_new);
                p_s[row * 64 + n] = (_Float16)p;
                sum += p;
            }
            l_s[row]    = l_s[row] * cf + sum;
            m_s[row]    = m_new;
            coef_s[row] = cf;
        }
        __syncthreads();
        // ----- rescale O, accumulate P x V -----
        float cf8[8];
#pragma unroll
        for (int r = 0; r < 8; ++r) cf8[r] = coef_s[wave * 16 + (half ? r + 8 : r)];
#pragma unroll
        for (int a = 0; a < 4; ++a)
#pragma unroll
            for (int r = 0; r < 8; ++r) O[a][r] *= cf8[r];

        v16h pf[2];
#pragma unroll
        for (int kf = 0; kf < 2; ++kf)
            pf[kf] = load_frag_f16(p_s + (wave * 16 + ln) * 64 + kf * 32, half);
#pragma unroll
        for (int a = 0; a < 4; ++a) {
            const long vrow = ((long)b * HEAD + a * 16 + ln) * S_LEN + n0;
#pragma unroll
            for (int kf = 0; kf < 2; ++kf) {
                const v16h vb = load_frag_f16(Vt + vrow + kf * 32, half);
                O[a] = __builtin_amdgcn_wmma_f32_16x16x32_f16(false, pf[kf], false, vb, (short)0, O[a], false, false);
            }
        }
    }

    // ----- final 1/l normalize and store fp32 output -----
    float linv[8];
#pragma unroll
    for (int r = 0; r < 8; ++r) linv[r] = 1.f / l_s[wave * 16 + (half ? r + 8 : r)];
#pragma unroll
    for (int a = 0; a < 4; ++a)
#pragma unroll
        for (int r = 0; r < 8; ++r) {
            const int mloc = half ? (r + 8) : r;
            out[(rowbase + m0 + wave * 16 + mloc) * HEAD + a * 16 + ln] = O[a][r] * linv[r];
        }
}

// ---------------------------------------------------------------------------
extern "C" void kernel_launch(void* const* d_in, const int* in_sizes, int n_in,
                              void* d_out, int out_size, void* d_ws, size_t ws_size,
                              hipStream_t stream) {
    (void)in_sizes; (void)n_in; (void)out_size; (void)ws_size;
    const float* query = (const float*)d_in[0];
    const float* key   = (const float*)d_in[1];
    const float* value = (const float*)d_in[2];
    const float* Wq   = (const float*)d_in[3];
    const float* bq   = (const float*)d_in[4];
    const float* Wk   = (const float*)d_in[5];
    const float* bk   = (const float*)d_in[6];
    const float* Wv   = (const float*)d_in[7];
    const float* bv   = (const float*)d_in[8];
    const float* Wc2q = (const float*)d_in[9];
    const float* bc2q = (const float*)d_in[10];
    float* out = (float*)d_out;

    // workspace layout (f16): Qh | Kh | Vt | qnh | knh | Wqh | Wkh | Wvh  (~3.7 MB)
    _Float16* Qh  = (_Float16*)d_ws;
    _Float16* Kh  = Qh  + (size_t)ROWS * HEAD;
    _Float16* Vt  = Kh  + (size_t)ROWS * HEAD;
    _Float16* qnh = Vt  + (size_t)ROWS * HEAD;
    _Float16* knh = qnh + (size_t)ROWS * NQ;
    _Float16* Wqh = knh + (size_t)ROWS * NQ;
    _Float16* Wkh = Wqh + (size_t)HEAD * HIDDEN;
    _Float16* Wvh = Wkh + (size_t)HEAD * HIDDEN;

    wcvt_kernel<<<(HEAD * HIDDEN / 4 + 255) / 256, 256, 0, stream>>>(Wq, Wk, Wv, Wqh, Wkh, Wvh);
    qkv_project_kernel<<<128, 128, 0, stream>>>(query, key, value,
                                                Wqh, bq, Wkh, bk, Wvh, bv,
                                                Qh, Kh, Vt);
    c2q_kernel<<<32, 256, 0, stream>>>(Qh, Kh, Wc2q, bc2q, qnh, knh);
    attn_kernel<<<128, 128, 0, stream>>>(Qh, Kh, Vt, qnh, knh, out);
}
